// GraphormerEncoder_25812753449660
// MI455X (gfx1250) — matrix operand order repaired and checked by reading, not compiled, not run on hardware
//
#include <hip/hip_runtime.h>
#include <hip/hip_bf16.h>
#include <math.h>

// ---------------- problem constants ----------------
static constexpr int B = 32, N = 256, NF = 3, NEF = 3;
static constexpr int D = 768, FF = 3072, H = 12, L = 6;
static constexpr int NA = 512, ND_ = 512, NE = 512, NS = 512;
static constexpr int S = N + 1;            // 257
static constexpr int HD = D / H;           // 64
static constexpr float SCALE = 0.125f;     // HD^-0.5
static constexpr float EPS = 1e-5f;
static constexpr int MT = B * S;           // 8224 token rows
static constexpr int SPAD = 288;           // S padded to 9*32 key tiles
static constexpr float BIGNEG = -1e30f;

typedef _Float16 v16h __attribute__((ext_vector_type(16)));
typedef _Float16 f16x8 __attribute__((ext_vector_type(8)));
typedef float    v8f  __attribute__((ext_vector_type(8)));
typedef _Float16 half_t;

// CDNA5 async global->LDS copies (ASYNCcnt) if the toolchain exposes them
#if defined(__has_builtin)
#if __has_builtin(__builtin_amdgcn_global_load_async_to_lds_b128) && \
    __has_builtin(__builtin_amdgcn_s_wait_asynccnt)
#define USE_ASYNC_LDS 1
#endif
#endif
#ifndef USE_ASYNC_LDS
#define USE_ASYNC_LDS 0
#endif

__device__ inline f16x8 zero8() {
  f16x8 z;
#pragma unroll
  for (int i = 0; i < 8; ++i) z[i] = (half_t)0;
  return z;
}
__device__ inline v8f zero8f() {
  v8f z;
#pragma unroll
  for (int i = 0; i < 8; ++i) z[i] = 0.f;
  return z;
}
__device__ inline v16h mk16(f16x8 lo, f16x8 hi) {
  v16h r;
#pragma unroll
  for (int i = 0; i < 8; ++i) { r[i] = lo[i]; r[8 + i] = hi[i]; }
  return r;
}
__device__ inline f16x8 ldg8(const half_t* p) { return *reinterpret_cast<const f16x8*>(p); }
__device__ inline int iclamp(int v, int lo, int hi) { return v < lo ? lo : (v > hi ? hi : v); }

#if USE_ASYNC_LDS
typedef int i32x4g __attribute__((vector_size(16)));
typedef __attribute__((address_space(1))) i32x4g gvec_t;
typedef __attribute__((address_space(3))) i32x4g lvec_t;
__device__ inline void async_cp16(const half_t* g, half_t* l) {
  __builtin_amdgcn_global_load_async_to_lds_b128(
      (gvec_t*)(uintptr_t)g,
      (lvec_t*)(unsigned int)(uintptr_t)l,
      0, 0);
}
#endif

// xor-reduce across the 16-lane halves of a wave32 (rows live in 16-lane groups)
__device__ inline float redmax16(float v) {
  v = fmaxf(v, __int_as_float(__builtin_amdgcn_ds_swizzle(__float_as_int(v), 0x041F)));
  v = fmaxf(v, __int_as_float(__builtin_amdgcn_ds_swizzle(__float_as_int(v), 0x081F)));
  v = fmaxf(v, __int_as_float(__builtin_amdgcn_ds_swizzle(__float_as_int(v), 0x101F)));
  v = fmaxf(v, __int_as_float(__builtin_amdgcn_ds_swizzle(__float_as_int(v), 0x201F)));
  return v;
}
__device__ inline float redsum16(float v) {
  v += __int_as_float(__builtin_amdgcn_ds_swizzle(__float_as_int(v), 0x041F));
  v += __int_as_float(__builtin_amdgcn_ds_swizzle(__float_as_int(v), 0x081F));
  v += __int_as_float(__builtin_amdgcn_ds_swizzle(__float_as_int(v), 0x101F));
  v += __int_as_float(__builtin_amdgcn_ds_swizzle(__float_as_int(v), 0x201F));
  return v;
}

// ---------------- fp32 -> fp16 convert ----------------
__global__ __launch_bounds__(256) void cvt_f16_kernel(const float* __restrict__ in,
                                                      half_t* __restrict__ out, int n) {
  int i = blockIdx.x * 256 + threadIdx.x;
  if (i < n) out[i] = (half_t)in[i];
}

// ---------------- embedding: build h (B,S,D) fp32 ----------------
__global__ __launch_bounds__(256) void embed_kernel(
    const int* __restrict__ x, const int* __restrict__ in_deg, const int* __restrict__ out_deg,
    const float* __restrict__ atom_emb, const float* __restrict__ ind_emb,
    const float* __restrict__ outd_emb, const float* __restrict__ gtok,
    float* __restrict__ h) {
  int row = blockIdx.x;            // b*S + s
  int b = row / S, s = row % S;
#pragma unroll
  for (int t = 0; t < 3; ++t) {
    int c = threadIdx.x + 256 * t;
    float v;
    if (s == 0) {
      v = gtok[c];
    } else {
      int s1 = s - 1;
      v = 0.f;
#pragma unroll
      for (int f = 0; f < NF; ++f) {
        int a = iclamp(x[(b * N + s1) * NF + f], 0, NA);
        v += atom_emb[(size_t)a * D + c];
      }
      v += ind_emb[(size_t)iclamp(in_deg[b * N + s1], 0, ND_ - 1) * D + c];
      v += outd_emb[(size_t)iclamp(out_deg[b * N + s1], 0, ND_ - 1) * D + c];
    }
    h[(size_t)row * D + c] = v;
  }
}

// ---------------- graph bias gb (B,H,S,S) fp32, key-mask folded in ----------------
__global__ __launch_bounds__(256) void bias_kernel(
    const float* __restrict__ attn_bias, const int* __restrict__ sp,
    const int* __restrict__ aet, const int* __restrict__ x,
    const float* __restrict__ sp_emb, const float* __restrict__ e_emb,
    const float* __restrict__ virt, float* __restrict__ gb) {
  long long idx = (long long)blockIdx.x * 256 + threadIdx.x;
  if (idx >= (long long)B * S * S) return;
  int k = (int)(idx % S); long long r = idx / S;
  int q = (int)(r % S); int b = (int)(r / S);
  bool masked = (k > 0) && (x[(b * N + (k - 1)) * NF + 0] == 0);
  float base = attn_bias[idx];
  int spv = 0, a0 = 0, a1 = 0, a2 = 0;
  bool innerv = (q > 0) && (k > 0);
  if (innerv) {
    size_t e = (size_t)(b * N + (q - 1)) * N + (k - 1);
    spv = iclamp(sp[e], 0, NS - 1);
    a0 = iclamp(aet[e * NEF + 0], 0, NE);
    a1 = iclamp(aet[e * NEF + 1], 0, NE);
    a2 = iclamp(aet[e * NEF + 2], 0, NE);
  }
  bool addv = (q == 0) || (k == 0);
#pragma unroll
  for (int hh = 0; hh < H; ++hh) {
    float v = base;
    if (innerv)
      v += sp_emb[spv * H + hh] +
           (e_emb[a0 * H + hh] + e_emb[a1 * H + hh] + e_emb[a2 * H + hh]) * (1.f / 3.f);
    if (addv) v += virt[hh];
    if (masked) v = BIGNEG;
    gb[(((size_t)(b * H + hh)) * S + q) * S + k] = v;
  }
}

// ---------------- LayerNorm -> f16 activations ----------------
__global__ __launch_bounds__(256) void ln_f16_kernel(const float* __restrict__ xin,
                                                     const float* __restrict__ g,
                                                     const float* __restrict__ bta,
                                                     half_t* __restrict__ out) {
  __shared__ float rs[256], rq[256];
  int row = blockIdx.x, tid = threadIdx.x;
  const float* xr = xin + (size_t)row * D;
  float v0 = xr[tid], v1 = xr[tid + 256], v2 = xr[tid + 512];
  rs[tid] = v0 + v1 + v2;
  rq[tid] = v0 * v0 + v1 * v1 + v2 * v2;
  __syncthreads();
  for (int off = 128; off; off >>= 1) {
    if (tid < off) { rs[tid] += rs[tid + off]; rq[tid] += rq[tid + off]; }
    __syncthreads();
  }
  float mean = rs[0] * (1.f / D);
  float var = rq[0] * (1.f / D) - mean * mean;
  float rstd = rsqrtf(var + EPS);
  half_t* orow = out + (size_t)row * D;
  orow[tid]       = (half_t)((v0 - mean) * rstd * g[tid]       + bta[tid]);
  orow[tid + 256] = (half_t)((v1 - mean) * rstd * g[tid + 256] + bta[tid + 256]);
  orow[tid + 512] = (half_t)((v2 - mean) * rstd * g[tid + 512] + bta[tid + 512]);
}

// ---------------- final LN + scatter to d_out ----------------
__global__ __launch_bounds__(256) void final_kernel(const float* __restrict__ xin,
                                                    const float* __restrict__ g,
                                                    const float* __restrict__ bta,
                                                    float* __restrict__ out) {
  __shared__ float rs[256], rq[256];
  int row = blockIdx.x, tid = threadIdx.x;
  int b = row / S, s = row % S;
  const float* xr = xin + (size_t)row * D;
  float v0 = xr[tid], v1 = xr[tid + 256], v2 = xr[tid + 512];
  rs[tid] = v0 + v1 + v2;
  rq[tid] = v0 * v0 + v1 * v1 + v2 * v2;
  __syncthreads();
  for (int off = 128; off; off >>= 1) {
    if (tid < off) { rs[tid] += rs[tid + off]; rq[tid] += rq[tid + off]; }
    __syncthreads();
  }
  float mean = rs[0] * (1.f / D);
  float rstd = rsqrtf(rq[0] * (1.f / D) - mean * mean + EPS);
  float* dst = (s == 0) ? (out + (size_t)B * N * D + (size_t)b * D)
                        : (out + ((size_t)(b * N + (s - 1))) * D);
  dst[tid]       = (v0 - mean) * rstd * g[tid]       + bta[tid];
  dst[tid + 256] = (v1 - mean) * rstd * g[tid + 256] + bta[tid + 256];
  dst[tid + 512] = (v2 - mean) * rstd * g[tid + 512] + bta[tid + 512];
}

// ---------------- WMMA GEMM: out[m,n] = A[m,:]·W[n,:] + bias[n] ----------------
// MODE 0: f16 store of (acc+bias)*alpha   (Q/K/V projections)
// MODE 1: f16 store of gelu(acc+bias)     (FFN up-proj)
// MODE 2: fp32 residual accumulate into outF (O-proj / FFN down-proj)
// Double-buffered LDS staging; async global->LDS (ASYNCcnt) when available.
template <int MODE>
__global__ __launch_bounds__(256) void gemm_kernel(
    const half_t* __restrict__ A, const half_t* __restrict__ W,
    const float* __restrict__ bias, float* __restrict__ outF,
    half_t* __restrict__ outH, int M, int Nn, int K, float alpha) {
  __shared__ __align__(16) half_t As[2][64][40];
  __shared__ __align__(16) half_t Bs[2][128][40];
  const int tid = threadIdx.x;
  const int wid = tid >> 5, lane = tid & 31;
  const int wm = wid & 1, wn = wid >> 1;
  const int lh = lane & 15;
  const int hi16 = (lane >= 16);
  const int m0 = blockIdx.x * 64;
  const int n0 = blockIdx.y * 128;

  v8f acc[2][2];
#pragma unroll
  for (int j = 0; j < 2; ++j)
#pragma unroll
    for (int i = 0; i < 2; ++i) acc[j][i] = zero8f();

  const int arow = tid >> 2, acg = (tid & 3) * 8;
  int gr = m0 + arow; if (gr >= M) gr = M - 1;   // clamp: padded rows discarded later
  const int nt = K / 32;

#if USE_ASYNC_LDS
  auto issue_tile = [&](int t, int buf) {
    const int kk = t * 32;
    async_cp16(A + (size_t)gr * K + kk + acg, &As[buf][arow][acg]);
#pragma unroll
    for (int p2 = 0; p2 < 2; ++p2) {
      int ch = tid * 2 + p2;
      int brow = ch >> 2, bcg = (ch & 3) * 8;
      async_cp16(W + (size_t)(n0 + brow) * K + kk + bcg, &Bs[buf][brow][bcg]);
    }
  };
  issue_tile(0, 0);
#endif

  for (int t = 0; t < nt; ++t) {
    const int buf = t & 1;
#if USE_ASYNC_LDS
    __builtin_amdgcn_s_wait_asynccnt(0);   // tile t resident in LDS
    __syncthreads();                       // all waves see it; prev reads of buf^1 done
    if (t + 1 < nt) issue_tile(t + 1, buf ^ 1);  // overlap next fetch with compute
#else
    const int kk = t * 32;
    {
      *reinterpret_cast<f16x8*>(&As[buf][arow][acg]) =
          ldg8(A + (size_t)gr * K + kk + acg);
    }
#pragma unroll
    for (int p2 = 0; p2 < 2; ++p2) {
      int ch = tid * 2 + p2;
      int brow = ch >> 2, bcg = (ch & 3) * 8;
      *reinterpret_cast<f16x8*>(&Bs[buf][brow][bcg]) =
          ldg8(W + (size_t)(n0 + brow) * K + kk + bcg);
    }
    __syncthreads();
#endif

    v16h af[2], bf[2];
#pragma unroll
    for (int j = 0; j < 2; ++j) {
      int r = 32 * wm + 16 * j + lh;
      int klo = hi16 ? 8 : 0;
      af[j] = mk16(*reinterpret_cast<const f16x8*>(&As[buf][r][klo]),
                   *reinterpret_cast<const f16x8*>(&As[buf][r][16 + klo]));
    }
#pragma unroll
    for (int i = 0; i < 2; ++i) {
      int nr = 32 * wn + 16 * i + lh;
      int koff = hi16 ? 8 : 0;
      bf[i] = mk16(*reinterpret_cast<const f16x8*>(&Bs[buf][nr][koff]),
                   *reinterpret_cast<const f16x8*>(&Bs[buf][nr][koff + 16]));
    }
#pragma unroll
    for (int j = 0; j < 2; ++j)
#pragma unroll
      for (int i = 0; i < 2; ++i)
        acc[j][i] = __builtin_amdgcn_wmma_f32_16x16x32_f16(
            false, af[j], false, bf[i], (short)0, acc[j][i], false, false);
#if !USE_ASYNC_LDS
    __syncthreads();
#endif
  }

#pragma unroll
  for (int j = 0; j < 2; ++j)
#pragma unroll
    for (int i = 0; i < 2; ++i) {
      int col = n0 + 32 * wn + 16 * i + lh;
      float bv = bias[col];
#pragma unroll
      for (int r = 0; r < 8; ++r) {
        int row = m0 + 32 * wm + 16 * j + r + (hi16 ? 8 : 0);
        if (row < M) {
          float v = acc[j][i][r] + bv;
          if (MODE == 0) {
            outH[(size_t)row * Nn + col] = (half_t)(v * alpha);
          } else if (MODE == 1) {
            float ge = 0.5f * v * (1.f + erff(v * 0.70710678118654752f));
            outH[(size_t)row * Nn + col] = (half_t)ge;
          } else {
            outF[(size_t)row * Nn + col] += v;
          }
        }
      }
    }
}

// ---------------- flash attention, one workgroup per (b,h) ----------------
__global__ __launch_bounds__(256) void attn_kernel(
    const half_t* __restrict__ Q, const half_t* __restrict__ Kb,
    const half_t* __restrict__ V, const float* __restrict__ gb,
    half_t* __restrict__ O) {
  __shared__ __align__(16) half_t VT[64][296];      // V transposed, zero-padded tokens
  __shared__ __align__(16) half_t Pb[8][16][40];    // per-wave P re-layout bounce
  const int b = blockIdx.x / H, hh = blockIdx.x % H;
  const int tid = threadIdx.x, wid = tid >> 5, lane = tid & 31;
  const int lh = lane & 15, hi16 = (lane >= 16);

  for (int idx = tid; idx < SPAD * 64; idx += 256) {
    int t = idx >> 6, d = idx & 63;
    VT[d][t] = (t < S) ? V[((size_t)(b * S + t)) * D + hh * HD + d] : (half_t)0;
  }
  __syncthreads();

  for (int qt = wid; qt < 17; qt += 8) {          // 17 q-tiles cover 272 >= S rows
    const int q0 = qt * 16;
    // Q fragments for d-chunks [0,32) and [32,64)
    v16h aq[2];
    {
      int rowq = q0 + lh; if (rowq >= S) rowq = S - 1;
      int klo = hi16 ? 8 : 0;
#pragma unroll
      for (int ds = 0; ds < 2; ++ds) {
        const half_t* p = Q + ((size_t)(b * S + rowq)) * D + hh * HD + 32 * ds;
        aq[ds] = mk16(ldg8(p + klo), ldg8(p + 16 + klo));
      }
    }
    float m_run[8], l_run[8];
#pragma unroll
    for (int j = 0; j < 8; ++j) { m_run[j] = -3.0e38f; l_run[j] = 0.f; }
    v8f accO[4];
#pragma unroll
    for (int t = 0; t < 4; ++t) accO[t] = zero8f();

    for (int kp = 0; kp < SPAD / 32; ++kp) {      // key-token pairs of 16
      // prefetch next key block rows into cache (global_prefetch_b8)
      {
        int ntok = kp * 32 + 32 + lane; if (ntok >= S) ntok = S - 1;
        __builtin_prefetch(Kb + ((size_t)(b * S + ntok)) * D + hh * HD, 0, 1);
      }
      v8f s[2] = { zero8f(), zero8f() };
#pragma unroll
      for (int half2 = 0; half2 < 2; ++half2) {
        int tok = kp * 32 + 16 * half2 + lh; if (tok >= S) tok = S - 1;
#pragma unroll
        for (int ds = 0; ds < 2; ++ds) {
          const half_t* p = Kb + ((size_t)(b * S + tok)) * D + hh * HD + 32 * ds + (hi16 ? 8 : 0);
          v16h bf = mk16(ldg8(p), ldg8(p + 16));
          s[half2] = __builtin_amdgcn_wmma_f32_16x16x32_f16(
              false, aq[ds], false, bf, (short)0, s[half2], false, false);
        }
      }
      // bias + pad-mask, online softmax update
      const int rbase = q0 + (hi16 ? 8 : 0);
      const int colk = kp * 32 + lh;
      float f[8];
#pragma unroll
      for (int j = 0; j < 8; ++j) {
        int row = rbase + j; if (row >= S) row = S - 1;
        const float* gbr = gb + (((size_t)(b * H + hh)) * S + row) * S;
        float s0 = (colk < S)      ? (s[0][j] + gbr[colk])      : BIGNEG;
        float s1 = (colk + 16 < S) ? (s[1][j] + gbr[colk + 16]) : BIGNEG;
        float t = redmax16(fmaxf(s0, s1));
        float mn = fmaxf(m_run[j], t);           // kp==0 contains col 0 -> finite
        f[j] = expf(m_run[j] - mn);
        float e0 = expf(s0 - mn), e1 = expf(s1 - mn);
        l_run[j] = l_run[j] * f[j] + redsum16(e0 + e1);
        m_run[j] = mn;
        int prow = j + (hi16 ? 8 : 0);
        Pb[wid][prow][lh]      = (half_t)e0;
        Pb[wid][prow][lh + 16] = (half_t)e1;
      }
#pragma unroll
      for (int t = 0; t < 4; ++t)
#pragma unroll
        for (int j = 0; j < 8; ++j) accO[t][j] *= f[j];
      // P tile as A fragment (16x32)
      v16h ap;
      {
        int klo = hi16 ? 8 : 0;
        ap = mk16(*reinterpret_cast<const f16x8*>(&Pb[wid][lh][klo]),
                  *reinterpret_cast<const f16x8*>(&Pb[wid][lh][16 + klo]));
      }
#pragma unroll
      for (int dt = 0; dt < 4; ++dt) {
        int tok0 = kp * 32 + (hi16 ? 8 : 0);
        int dcol = 16 * dt + lh;
        v16h bf = mk16(*reinterpret_cast<const f16x8*>(&VT[dcol][tok0]),
                       *reinterpret_cast<const f16x8*>(&VT[dcol][tok0 + 16]));
        accO[dt] = __builtin_amdgcn_wmma_f32_16x16x32_f16(
            false, ap, false, bf, (short)0, accO[dt], false, false);
      }
    }
#pragma unroll
    for (int dt = 0; dt < 4; ++dt)
#pragma unroll
      for (int j = 0; j < 8; ++j) {
        int row = q0 + j + (hi16 ? 8 : 0);
        if (row < S)
          O[((size_t)(b * S + row)) * D + hh * HD + 16 * dt + lh] =
              (half_t)(accO[dt][j] / l_run[j]);
      }
  }
}

// ---------------- host orchestration ----------------
extern "C" void kernel_launch(void* const* d_in, const int* in_sizes, int n_in,
                              void* d_out, int out_size, void* d_ws, size_t ws_size,
                              hipStream_t stream) {
  const int*   x        = (const int*)d_in[0];
  const float* attnb    = (const float*)d_in[1];
  const int*   sp       = (const int*)d_in[2];
  const int*   aet      = (const int*)d_in[3];
  const int*   indeg    = (const int*)d_in[4];
  const int*   outdeg   = (const int*)d_in[5];
  const float* atom_emb = (const float*)d_in[6];
  const float* ind_emb  = (const float*)d_in[7];
  const float* outd_emb = (const float*)d_in[8];
  const float* gtok     = (const float*)d_in[9];
  const float* sp_emb   = (const float*)d_in[10];
  const float* e_emb    = (const float*)d_in[11];
  const float* virt     = (const float*)d_in[12];
  const float* Wq = (const float*)d_in[13];
  const float* Wk = (const float*)d_in[14];
  const float* Wv = (const float*)d_in[15];
  const float* Wo = (const float*)d_in[16];
  const float* bq = (const float*)d_in[17];
  const float* bk = (const float*)d_in[18];
  const float* bv = (const float*)d_in[19];
  const float* bo = (const float*)d_in[20];
  const float* ln1g = (const float*)d_in[21];
  const float* ln1b = (const float*)d_in[22];
  const float* ln2g = (const float*)d_in[23];
  const float* ln2b = (const float*)d_in[24];
  const float* W1 = (const float*)d_in[25];
  const float* b1 = (const float*)d_in[26];
  const float* W2 = (const float*)d_in[27];
  const float* b2 = (const float*)d_in[28];
  const float* fing = (const float*)d_in[29];
  const float* finb = (const float*)d_in[30];

  // workspace layout (all 16B aligned)
  char* p = (char*)d_ws;
  float* gbuf = (float*)p;           p += (size_t)B * H * S * S * 4;     // 101.5 MB
  float* hbuf = (float*)p;           p += (size_t)MT * D * 4;            // 25.3 MB
  half_t* z16 = (half_t*)p;          p += (size_t)MT * D * 2;
  half_t* q16 = (half_t*)p;          p += (size_t)MT * D * 2;
  half_t* k16 = (half_t*)p;          p += (size_t)MT * D * 2;
  half_t* v16 = (half_t*)p;          p += (size_t)MT * D * 2;
  half_t* o16 = (half_t*)p;          p += (size_t)MT * D * 2;
  half_t* f16b = (half_t*)p;         p += (size_t)MT * FF * 2;           // 50.5 MB
  half_t* wbuf = (half_t*)p;         p += (size_t)FF * D * 2;            // 4.7 MB (reused)

  // embeddings + graph bias (layer-invariant)
  embed_kernel<<<MT, 256, 0, stream>>>(x, indeg, outdeg, atom_emb, ind_emb, outd_emb,
                                       gtok, hbuf);
  {
    long long tot = (long long)B * S * S;
    bias_kernel<<<(int)((tot + 255) / 256), 256, 0, stream>>>(attnb, sp, aet, x, sp_emb,
                                                              e_emb, virt, gbuf);
  }

  dim3 gProj((MT + 63) / 64, D / 128);
  dim3 gFF1((MT + 63) / 64, FF / 128);
  const int nDD = D * D, nFD = FF * D;
  auto cvt = [&](const float* src, int n) {
    cvt_f16_kernel<<<(n + 255) / 256, 256, 0, stream>>>(src, wbuf, n);
  };

  for (int l = 0; l < L; ++l) {
    // --- attention block ---
    ln_f16_kernel<<<MT, 256, 0, stream>>>(hbuf, ln1g + l * D, ln1b + l * D, z16);
    cvt(Wq + (size_t)l * nDD, nDD);
    gemm_kernel<0><<<gProj, 256, 0, stream>>>(z16, wbuf, bq + l * D, nullptr, q16,
                                              MT, D, D, SCALE);
    cvt(Wk + (size_t)l * nDD, nDD);
    gemm_kernel<0><<<gProj, 256, 0, stream>>>(z16, wbuf, bk + l * D, nullptr, k16,
                                              MT, D, D, 1.f);
    cvt(Wv + (size_t)l * nDD, nDD);
    gemm_kernel<0><<<gProj, 256, 0, stream>>>(z16, wbuf, bv + l * D, nullptr, v16,
                                              MT, D, D, 1.f);
    attn_kernel<<<B * H, 256, 0, stream>>>(q16, k16, v16, gbuf, o16);
    cvt(Wo + (size_t)l * nDD, nDD);
    gemm_kernel<2><<<gProj, 256, 0, stream>>>(o16, wbuf, bo + l * D, hbuf, nullptr,
                                              MT, D, D, 1.f);
    // --- FFN block ---
    ln_f16_kernel<<<MT, 256, 0, stream>>>(hbuf, ln2g + l * D, ln2b + l * D, z16);
    cvt(W1 + (size_t)l * nFD, nFD);
    gemm_kernel<1><<<gFF1, 256, 0, stream>>>(z16, wbuf, b1 + l * FF, nullptr, f16b,
                                             MT, FF, D, 1.f);
    cvt(W2 + (size_t)l * nFD, nFD);
    gemm_kernel<2><<<gProj, 256, 0, stream>>>(f16b, wbuf, b2 + l * D, hbuf, nullptr,
                                              MT, D, FF, 1.f);
  }

  final_kernel<<<MT, 256, 0, stream>>>(hbuf, fing, finb, (float*)d_out);
}
